// TransformerDecoderLayer_38714835206639
// MI455X (gfx1250) — matrix operand order repaired and checked by reading, not compiled, not run on hardware
//
#include <hip/hip_runtime.h>
#include <hip/hip_bf16.h>

// ---------------------------------------------------------------------------
// MI455X (gfx1250) DAB-DETR TransformerDecoderLayer forward.
// All matmuls run on v_wmma_f32_16x16x32_bf16 (wave32 WMMA, f32 accumulate).
// ---------------------------------------------------------------------------

typedef __attribute__((ext_vector_type(16))) __bf16 v16bf;
typedef __attribute__((ext_vector_type(8)))  __bf16 v8bf;
typedef __attribute__((ext_vector_type(8)))  float  v8f;

union ABfrag { v16bf v; struct { v8bf lo; v8bf hi; } h; };

__device__ __forceinline__ v8f wmma_bf16(v16bf a, v16bf b, v8f c) {
  // (neg_a, A, neg_b, B, c_mod, C, reuse_a, reuse_b)
  return __builtin_amdgcn_wmma_f32_16x16x32_bf16(false, a, false, b, (short)0, c,
                                                 false, false);
}

__device__ __forceinline__ float negInf() { return -__builtin_inff(); }

// --------------------------- fp32 -> bf16 convert ---------------------------
__global__ void k_cvt(const float* __restrict__ in, __bf16* __restrict__ out,
                      size_t n) {
  size_t i = (size_t)blockIdx.x * blockDim.x + threadIdx.x;
  size_t stride = (size_t)gridDim.x * blockDim.x;
  for (; i < n; i += stride) out[i] = (__bf16)in[i];
}

// ------------------------------- GEMM (WMMA) --------------------------------
// Y[M,N] = ((X1 @ W1^T + b1) [+ (X2 @ W2^T + b2)]) * scale [+ resid] [relu]
// torch convention: W is [N,K] row-major -> row n of W is the K-contiguous
// column n of B = W^T, ideal for the WMMA B fragment (one v16bf load).
// One wave computes a 32x32 tile; block = 256 threads = 8 waves covering
// 8 consecutive N tiles. Requires M%32==0, N%256==0, K%32==0 (true here).
// PAIR2 is a compile-time flag so the K-loop is branch-free straight-line code.
template <bool PAIR2, bool RELU>
__global__ void k_gemm(const __bf16* __restrict__ X1, const __bf16* __restrict__ W1,
                       const float* __restrict__ b1,
                       const __bf16* __restrict__ X2, const __bf16* __restrict__ W2,
                       const float* __restrict__ b2,
                       const float* __restrict__ resid,
                       float* __restrict__ outf, __bf16* __restrict__ outbf,
                       int M, int N, int K, float scale) {
  const int wv = threadIdx.x >> 5;
  const int lane = threadIdx.x & 31;
  const int grp = lane >> 4;      // lane half: selects K/M sub-block
  const int ln = lane & 15;
  const int n0 = (blockIdx.x * 8 + wv) * 32;
  const int m0 = blockIdx.y * 32;

  v8f acc[2][2] = {};
  for (int k0 = 0; k0 < K; k0 += 32) {
    ABfrag a[2], bb[2];
#pragma unroll
    for (int i = 0; i < 2; ++i) {
      const __bf16* xb = X1 + (size_t)(m0 + i * 16 + ln) * K + k0;
      a[i].h.lo = *(const v8bf*)(xb + grp * 16);
      a[i].h.hi = *(const v8bf*)(xb + grp * 16 + 8);
      const __bf16* wb = W1 + (size_t)(n0 + i * 16 + ln) * K + k0;
      bb[i].v = *(const v16bf*)(wb + grp * 16);
    }
#pragma unroll
    for (int i = 0; i < 2; ++i)
#pragma unroll
      for (int j = 0; j < 2; ++j) acc[i][j] = wmma_bf16(a[i].v, bb[j].v, acc[i][j]);

    if constexpr (PAIR2) {  // fused second input pair (content + positional)
      ABfrag a2[2], bb2[2];
#pragma unroll
      for (int i = 0; i < 2; ++i) {
        const __bf16* xb = X2 + (size_t)(m0 + i * 16 + ln) * K + k0;
        a2[i].h.lo = *(const v8bf*)(xb + grp * 16);
        a2[i].h.hi = *(const v8bf*)(xb + grp * 16 + 8);
        const __bf16* wb = W2 + (size_t)(n0 + i * 16 + ln) * K + k0;
        bb2[i].v = *(const v16bf*)(wb + grp * 16);
      }
#pragma unroll
      for (int i = 0; i < 2; ++i)
#pragma unroll
        for (int j = 0; j < 2; ++j)
          acc[i][j] = wmma_bf16(a2[i].v, bb2[j].v, acc[i][j]);
    }
  }

#pragma unroll
  for (int i = 0; i < 2; ++i)
#pragma unroll
    for (int j = 0; j < 2; ++j)
#pragma unroll
      for (int r = 0; r < 8; ++r) {
        const int row = m0 + i * 16 + grp * 8 + r;   // C/D layout: M in VGPRs
        const int col = n0 + j * 16 + ln;            // N across lanes
        float v = acc[i][j][r] + b1[col];
        if constexpr (PAIR2) v += b2[col];
        v *= scale;
        const size_t o = (size_t)row * N + col;
        if (resid) v += resid[o];
        if constexpr (RELU) v = fmaxf(v, 0.f);
        if (outf) outf[o] = v;
        if (outbf) outbf[o] = (__bf16)v;
      }
}

// ----------------------- attention scores + softmax -------------------------
// Per (b,h,l-tile of 16): two-pass (flash-style) online softmax.
// scores[l,s] = q1[l]·k1[s] (+ q2[l]·k2[s])  -- each K=32 WMMA accumulation.
// q/k tensors are [seq*16+b, 256] bf16, per-head slice at h*32.
// Writes normalized bf16 weights to Wbuf[(b*8+h)*Lp + l, s] with zero padding.
template <bool HAS2>
__global__ void k_attn_softmax(const __bf16* __restrict__ Q1, const __bf16* __restrict__ K1,
                               const __bf16* __restrict__ Q2, const __bf16* __restrict__ K2,
                               __bf16* __restrict__ Wbuf, int L, int S, int Lp, int Sp) {
  const int Bsz = 16, D = 256, HD = 32;
  const int lt = blockIdx.x, h = blockIdx.y, b = blockIdx.z;
  const int wv = threadIdx.x >> 5;
  const int lane = threadIdx.x & 31;
  const int grp = lane >> 4, ln = lane & 15;
  const int l0 = lt * 16;
  const int nst = Sp / 16;

  // A fragments (rows = query rows, clamped; invalid rows masked at the end)
  const int lrow = min(l0 + ln, L - 1);
  ABfrag a1, a2;
  {
    const __bf16* qb = Q1 + ((size_t)(lrow * Bsz + b)) * D + h * HD;
    a1.h.lo = *(const v8bf*)(qb + grp * 16);
    a1.h.hi = *(const v8bf*)(qb + grp * 16 + 8);
  }
  if constexpr (HAS2) {
    const __bf16* qb = Q2 + ((size_t)(lrow * Bsz + b)) * D + h * HD;
    a2.h.lo = *(const v8bf*)(qb + grp * 16);
    a2.h.hi = *(const v8bf*)(qb + grp * 16 + 8);
  }

  // ---- pass 1: running (max, sum) per row, this lane's column subset ----
  float rm[8], rs[8];
#pragma unroll
  for (int r = 0; r < 8; ++r) { rm[r] = negInf(); rs[r] = 0.f; }

  for (int st = wv; st < nst; st += 8) {
    const int s = st * 16 + ln;
    const int srow = min(s, S - 1);
    ABfrag bf1;
    bf1.v = *(const v16bf*)(K1 + ((size_t)(srow * Bsz + b)) * D + h * HD + grp * 16);
    v8f acc = {};
    acc = wmma_bf16(a1.v, bf1.v, acc);
    if constexpr (HAS2) {
      ABfrag bf2;
      bf2.v = *(const v16bf*)(K2 + ((size_t)(srow * Bsz + b)) * D + h * HD + grp * 16);
      acc = wmma_bf16(a2.v, bf2.v, acc);
    }
    if (s < S) {
#pragma unroll
      for (int r = 0; r < 8; ++r) {
        const float v = acc[r];
        const float mo = rm[r];
        const float mn = fmaxf(mo, v);
        rs[r] = rs[r] * __expf(mo - mn) + __expf(v - mn);
        rm[r] = mn;
      }
    }
  }
  // merge across the 16 lanes that share this row group (xor masks stay in half)
#pragma unroll
  for (int r = 0; r < 8; ++r) {
    float m = rm[r], ssum = rs[r];
    for (int mask = 1; mask < 16; mask <<= 1) {
      const float mo = __shfl_xor(m, mask, 32);
      const float so = __shfl_xor(ssum, mask, 32);
      const float mn = fmaxf(m, mo);
      const float e1 = (m > negInf()) ? ssum * __expf(m - mn) : 0.f;
      const float e2 = (mo > negInf()) ? so * __expf(mo - mn) : 0.f;
      m = mn; ssum = e1 + e2;
    }
    rm[r] = m; rs[r] = ssum;
  }
  __shared__ float sm[8][16], ss[8][16], fm[16], fs[16];
  if (ln == 0) {  // lanes 0 (rows 0-7) and 16 (rows 8-15)
#pragma unroll
    for (int r = 0; r < 8; ++r) { sm[wv][grp * 8 + r] = rm[r]; ss[wv][grp * 8 + r] = rs[r]; }
  }
  __syncthreads();
  if (threadIdx.x < 16) {
    const int r = threadIdx.x;
    float m = negInf(), ssum = 0.f;
    for (int w2 = 0; w2 < 8; ++w2) {
      const float mo = sm[w2][r], so = ss[w2][r];
      const float mn = fmaxf(m, mo);
      const float e1 = (m > negInf()) ? ssum * __expf(m - mn) : 0.f;
      const float e2 = (mo > negInf()) ? so * __expf(mo - mn) : 0.f;
      m = mn; ssum = e1 + e2;
    }
    fm[r] = m; fs[r] = ssum;
  }
  __syncthreads();

  // ---- pass 2: recompute scores, emit normalized bf16 weights ----
  const size_t wbase = ((size_t)(b * 8 + h)) * Lp * Sp;
  for (int st = wv; st < nst; st += 8) {
    const int s = st * 16 + ln;
    const int srow = min(s, S - 1);
    ABfrag bf1;
    bf1.v = *(const v16bf*)(K1 + ((size_t)(srow * Bsz + b)) * D + h * HD + grp * 16);
    v8f acc = {};
    acc = wmma_bf16(a1.v, bf1.v, acc);
    if constexpr (HAS2) {
      ABfrag bf2;
      bf2.v = *(const v16bf*)(K2 + ((size_t)(srow * Bsz + b)) * D + h * HD + grp * 16);
      acc = wmma_bf16(a2.v, bf2.v, acc);
    }
#pragma unroll
    for (int r = 0; r < 8; ++r) {
      const int row = grp * 8 + r;
      const int l = l0 + row;
      float w = 0.f;
      if (s < S && l < L) w = __expf(acc[r] - fm[row]) / fs[row];
      Wbuf[wbase + (size_t)(l0 + row) * Sp + s] = (__bf16)w;
    }
  }
}

// --------------------- V transpose to [b,h,c, Sp] layout --------------------
__global__ void k_transpose_v(const __bf16* __restrict__ V, __bf16* __restrict__ Vt,
                              int S, int Sp) {
  const size_t total = (size_t)16 * 8 * 32 * Sp;
  size_t i = (size_t)blockIdx.x * blockDim.x + threadIdx.x;
  const size_t stride = (size_t)gridDim.x * blockDim.x;
  for (; i < total; i += stride) {
    const int s = (int)(i % Sp);
    size_t t = i / Sp;
    const int c = (int)(t % 32); t /= 32;
    const int h = (int)(t % 8);
    const int b = (int)(t / 8);
    __bf16 v = (__bf16)0.f;
    if (s < S) v = V[((size_t)(s * 16 + b)) * 256 + h * 32 + c];
    Vt[i] = v;
  }
}

// -------------------------- attention-weights x V ---------------------------
// One wave per (b,h,l-tile): O[16,32] += W[16,Sp] @ V[Sp,32] (K steps of 32).
__global__ void k_av(const __bf16* __restrict__ Wbuf, const __bf16* __restrict__ Vt,
                     __bf16* __restrict__ O, int L, int Lp, int Sp) {
  const int Bsz = 16, D = 256, HD = 32;
  const int lt = blockIdx.x, h = blockIdx.y, b = blockIdx.z;
  const int lane = threadIdx.x & 31;
  const int grp = lane >> 4, ln = lane & 15;
  const int l0 = lt * 16;
  const size_t wbase = ((size_t)(b * 8 + h)) * Lp * Sp;
  const size_t vbase = ((size_t)(b * 8 + h)) * 32 * Sp;

  v8f acc0 = {}, acc1 = {};
  for (int kb = 0; kb < Sp; kb += 32) {
    ABfrag a;
    const __bf16* ab = Wbuf + wbase + (size_t)(l0 + ln) * Sp + kb;
    a.h.lo = *(const v8bf*)(ab + grp * 16);
    a.h.hi = *(const v8bf*)(ab + grp * 16 + 8);
    ABfrag b0, b1;
    b0.v = *(const v16bf*)(Vt + vbase + (size_t)(0 + ln) * Sp + kb + grp * 16);
    b1.v = *(const v16bf*)(Vt + vbase + (size_t)(16 + ln) * Sp + kb + grp * 16);
    acc0 = wmma_bf16(a.v, b0.v, acc0);
    acc1 = wmma_bf16(a.v, b1.v, acc1);
  }
#pragma unroll
  for (int r = 0; r < 8; ++r) {
    const int l = l0 + grp * 8 + r;
    if (l < L) {
      const size_t o = ((size_t)(l * Bsz + b)) * D + h * HD;
      O[o + ln] = (__bf16)acc0[r];
      O[o + 16 + ln] = (__bf16)acc1[r];
    }
  }
}

// -------------------------------- layernorm ---------------------------------
// One wave per 256-wide row; shfl-xor reductions (wave32).
__global__ void k_layernorm(const float* __restrict__ x, const float* __restrict__ g,
                            const float* __restrict__ be, float* __restrict__ outf,
                            __bf16* __restrict__ outbf) {
  const int row = blockIdx.x;
  const int lane = threadIdx.x;
  const float* xr = x + (size_t)row * 256;
  float v[8];
  float s = 0.f;
#pragma unroll
  for (int i = 0; i < 8; ++i) { v[i] = xr[lane + 32 * i]; s += v[i]; }
  for (int m = 1; m < 32; m <<= 1) s += __shfl_xor(s, m, 32);
  const float mean = s * (1.f / 256.f);
  float q = 0.f;
#pragma unroll
  for (int i = 0; i < 8; ++i) { const float d = v[i] - mean; q += d * d; }
  for (int m = 1; m < 32; m <<= 1) q += __shfl_xor(q, m, 32);
  const float inv = rsqrtf(q * (1.f / 256.f) + 1e-5f);
#pragma unroll
  for (int i = 0; i < 8; ++i) {
    const int e = lane + 32 * i;
    const float y = (v[i] - mean) * inv * g[e] + be[e];
    if (outf) outf[(size_t)row * 256 + e] = y;
    if (outbf) outbf[(size_t)row * 256 + e] = (__bf16)y;
  }
}

// ------------------------- head-averaged weights out ------------------------
__global__ void k_mean_heads(const __bf16* __restrict__ Wbuf, float* __restrict__ out,
                             int L, int S, int Lp, int Sp) {
  const size_t total = (size_t)16 * L * S;
  size_t i = (size_t)blockIdx.x * blockDim.x + threadIdx.x;
  const size_t stride = (size_t)gridDim.x * blockDim.x;
  for (; i < total; i += stride) {
    const int s = (int)(i % S);
    size_t t = i / S;
    const int l = (int)(t % L);
    const int b = (int)(t / L);
    float acc = 0.f;
    for (int h = 0; h < 8; ++h)
      acc += (float)Wbuf[(((size_t)(b * 8 + h)) * Lp + l) * Sp + s];
    out[i] = acc * 0.125f;
  }
}

// ----------------------------------------------------------------------------
extern "C" void kernel_launch(void* const* d_in, const int* in_sizes, int n_in,
                              void* d_out, int out_size, void* d_ws, size_t ws_size,
                              hipStream_t stream) {
  (void)in_sizes; (void)n_in; (void)out_size; (void)ws_size;
  constexpr int L = 300, Bz = 16, Sc = 2048, D = 256, FF = 2048;
  constexpr int Mq = L * Bz;       // 4800
  constexpr int Mm = Sc * Bz;      // 32768
  constexpr int Lp = 304;          // L padded to 16
  constexpr int SpS = 320;         // self S padded to 32
  constexpr int SpC = 2048;        // cross S (already multiple of 32)
  const float sc_self = 0.17677669529663687f;  // 1/sqrt(32)
  const float sc_cross = 0.125f;               // 1/sqrt(64)

  const float* tgt    = (const float*)d_in[0];
  const float* memory = (const float*)d_in[1];
  const float* pos    = (const float*)d_in[2];
  const float* qpos   = (const float*)d_in[3];
  const float* sine   = (const float*)d_in[4];
  // params: 13 (w,b) d->d, linear1, linear2, norm1..3 (insertion order)
  auto Wp = [&](int i) { return (const float*)d_in[5 + 2 * i]; };
  auto Bp = [&](int i) { return (const float*)d_in[6 + 2 * i]; };
  // idx: 0 sa_qc 1 sa_qp 2 sa_kc 3 sa_kp 4 sa_v 5 ca_qc 6 ca_qp 7 ca_kc
  //      8 ca_kp 9 ca_v 10 ca_sine 11 self_out 12 cross_out 13 lin1 14 lin2
  const float* g1 = (const float*)d_in[35]; const float* n1b = (const float*)d_in[36];
  const float* g2 = (const float*)d_in[37]; const float* n2b = (const float*)d_in[38];
  const float* g3 = (const float*)d_in[39]; const float* n3b = (const float*)d_in[40];

  float* out_tgt = (float*)d_out;
  float* out_qw  = out_tgt + (size_t)Mq * D;              // [16,300,300]
  float* out_cw  = out_qw + (size_t)Bz * L * L;           // [16,300,2048]

  // ---- workspace bump allocator ----
  char* wsp = (char*)d_ws;
  size_t off = 0;
  auto alloc = [&](size_t bytes) -> void* {
    void* p = wsp + off;
    off = (off + bytes + 255) & ~(size_t)255;
    return p;
  };
  auto cvt = [&](const float* src, __bf16* dst, size_t n) {
    const int blocks = (int)((n + 255) / 256);
    k_cvt<<<blocks, 256, 0, stream>>>(src, dst, n);
  };

  // bf16 copies of inputs
  __bf16* tgt_bf  = (__bf16*)alloc((size_t)Mq * D * 2);
  __bf16* qpos_bf = (__bf16*)alloc((size_t)Mq * D * 2);
  __bf16* sine_bf = (__bf16*)alloc((size_t)Mq * D * 2);
  __bf16* mem_bf  = (__bf16*)alloc((size_t)Mm * D * 2);
  __bf16* pos_bf  = (__bf16*)alloc((size_t)Mm * D * 2);
  cvt(tgt, tgt_bf, (size_t)Mq * D);
  cvt(qpos, qpos_bf, (size_t)Mq * D);
  cvt(sine, sine_bf, (size_t)Mq * D);
  cvt(memory, mem_bf, (size_t)Mm * D);
  cvt(pos, pos_bf, (size_t)Mm * D);

  // bf16 copies of weights
  __bf16* wbf[15];
  for (int i = 0; i < 15; ++i) {
    const size_t n = (i == 13) ? (size_t)FF * D : (i == 14) ? (size_t)D * FF
                                                            : (size_t)D * D;
    wbf[i] = (__bf16*)alloc(n * 2);
    cvt(Wp(i), wbf[i], n);
  }

  // intermediates
  __bf16* q_s   = (__bf16*)alloc((size_t)Mq * D * 2);
  __bf16* k_s   = (__bf16*)alloc((size_t)Mq * D * 2);
  __bf16* v_s   = (__bf16*)alloc((size_t)Mq * D * 2);
  __bf16* Wself = (__bf16*)alloc((size_t)Bz * 8 * Lp * SpS * 2);
  __bf16* VtS   = (__bf16*)alloc((size_t)Bz * 8 * 32 * SpS * 2);
  __bf16* ao_s  = (__bf16*)alloc((size_t)Mq * D * 2);
  float*  ybuf  = (float*)alloc((size_t)Mq * D * 4);
  float*  tgt1f = (float*)alloc((size_t)Mq * D * 4);
  __bf16* tgt1b = (__bf16*)alloc((size_t)Mq * D * 2);
  __bf16* q_c   = (__bf16*)alloc((size_t)Mq * D * 2);
  __bf16* sin_p = (__bf16*)alloc((size_t)Mq * D * 2);
  __bf16* kc2   = (__bf16*)alloc((size_t)Mm * D * 2);
  __bf16* kp    = (__bf16*)alloc((size_t)Mm * D * 2);
  __bf16* v_c   = (__bf16*)alloc((size_t)Mm * D * 2);
  __bf16* Wcross = (__bf16*)alloc((size_t)Bz * 8 * Lp * SpC * 2);
  __bf16* VtC   = (__bf16*)alloc((size_t)Bz * 8 * 32 * SpC * 2);
  __bf16* ao_c  = (__bf16*)alloc((size_t)Mq * D * 2);
  float*  tgt2f = (float*)alloc((size_t)Mq * D * 4);
  __bf16* tgt2b = (__bf16*)alloc((size_t)Mq * D * 2);
  __bf16* h1    = (__bf16*)alloc((size_t)Mq * FF * 2);

  const dim3 blk256(256);
  auto gemm_grid = [](int M, int N) { return dim3(N / 256, M / 32); };

  // ================= self-attention =================
  // q = (tgt@Wqc^T+bqc + qpos@Wqp^T+bqp) * scale ; k = content+pos ; v
  k_gemm<true, false><<<gemm_grid(Mq, D), blk256, 0, stream>>>(tgt_bf, wbf[0], Bp(0),
      qpos_bf, wbf[1], Bp(1), nullptr, nullptr, q_s, Mq, D, D, sc_self);
  k_gemm<true, false><<<gemm_grid(Mq, D), blk256, 0, stream>>>(tgt_bf, wbf[2], Bp(2),
      qpos_bf, wbf[3], Bp(3), nullptr, nullptr, k_s, Mq, D, D, 1.f);
  k_gemm<false, false><<<gemm_grid(Mq, D), blk256, 0, stream>>>(tgt_bf, wbf[4], Bp(4),
      nullptr, nullptr, nullptr, nullptr, nullptr, v_s, Mq, D, D, 1.f);
  k_attn_softmax<false><<<dim3(Lp / 16, 8, Bz), blk256, 0, stream>>>(
      q_s, k_s, nullptr, nullptr, Wself, L, L, Lp, SpS);
  k_transpose_v<<<2048, 256, 0, stream>>>(v_s, VtS, L, SpS);
  k_av<<<dim3(Lp / 16, 8, Bz), dim3(32), 0, stream>>>(Wself, VtS, ao_s, L, Lp, SpS);
  // out-proj + residual(tgt), then LN1
  k_gemm<false, false><<<gemm_grid(Mq, D), blk256, 0, stream>>>(ao_s, wbf[11], Bp(11),
      nullptr, nullptr, nullptr, tgt, ybuf, nullptr, Mq, D, D, 1.f);
  k_layernorm<<<Mq, dim3(32), 0, stream>>>(ybuf, g1, n1b, tgt1f, tgt1b);

  // ================= cross-attention =================
  k_gemm<true, false><<<gemm_grid(Mq, D), blk256, 0, stream>>>(tgt1b, wbf[5], Bp(5),
      qpos_bf, wbf[6], Bp(6), nullptr, nullptr, q_c, Mq, D, D, sc_cross);
  k_gemm<false, false><<<gemm_grid(Mq, D), blk256, 0, stream>>>(sine_bf, wbf[10], Bp(10),
      nullptr, nullptr, nullptr, nullptr, nullptr, sin_p, Mq, D, D, sc_cross);
  k_gemm<true, false><<<gemm_grid(Mm, D), blk256, 0, stream>>>(mem_bf, wbf[7], Bp(7),
      pos_bf, wbf[8], Bp(8), nullptr, nullptr, kc2, Mm, D, D, 1.f);
  k_gemm<false, false><<<gemm_grid(Mm, D), blk256, 0, stream>>>(pos_bf, wbf[8], Bp(8),
      nullptr, nullptr, nullptr, nullptr, nullptr, kp, Mm, D, D, 1.f);
  k_gemm<false, false><<<gemm_grid(Mm, D), blk256, 0, stream>>>(mem_bf, wbf[9], Bp(9),
      nullptr, nullptr, nullptr, nullptr, nullptr, v_c, Mm, D, D, 1.f);
  // concat trick: scores = q_c·kc2 + sine·kp (two K=32 WMMA accumulations)
  k_attn_softmax<true><<<dim3(Lp / 16, 8, Bz), blk256, 0, stream>>>(
      q_c, kc2, sin_p, kp, Wcross, L, Sc, Lp, SpC);
  k_transpose_v<<<8192, 256, 0, stream>>>(v_c, VtC, Sc, SpC);
  k_av<<<dim3(Lp / 16, 8, Bz), dim3(32), 0, stream>>>(Wcross, VtC, ao_c, L, Lp, SpC);
  k_gemm<false, false><<<gemm_grid(Mq, D), blk256, 0, stream>>>(ao_c, wbf[12], Bp(12),
      nullptr, nullptr, nullptr, tgt1f, ybuf, nullptr, Mq, D, D, 1.f);
  k_layernorm<<<Mq, dim3(32), 0, stream>>>(ybuf, g2, n2b, tgt2f, tgt2b);

  // ================= FFN =================
  k_gemm<false, true><<<gemm_grid(Mq, FF), blk256, 0, stream>>>(tgt2b, wbf[13], Bp(13),
      nullptr, nullptr, nullptr, nullptr, nullptr, h1, Mq, FF, D, 1.f);
  k_gemm<false, false><<<gemm_grid(Mq, D), blk256, 0, stream>>>(h1, wbf[14], Bp(14),
      nullptr, nullptr, nullptr, tgt2f, ybuf, nullptr, Mq, D, FF, 1.f);
  k_layernorm<<<Mq, dim3(32), 0, stream>>>(ybuf, g3, n3b, out_tgt, nullptr);

  // ================= averaged attention weights =================
  k_mean_heads<<<8192, 256, 0, stream>>>(Wself, out_qw, L, L, Lp, SpS);
  k_mean_heads<<<38400, 256, 0, stream>>>(Wcross, out_cw, L, Sc, Lp, SpC);
}
